// DifferentiableCGCNN_31559419691862
// MI455X (gfx1250) — matrix operand order repaired and checked by reading, not compiled, not run on hardware
//
#include <hip/hip_runtime.h>
#include <hip/hip_bf16.h>

typedef _Float16 v16h __attribute__((ext_vector_type(16)));
typedef float    v8f  __attribute__((ext_vector_type(8)));

#define NATOMS   4096
#define SPECIES  100
#define FEA      64
#define MAXN     12
#define KDIM     192            // 2*FEA + 64 gaussians
#define NOUT     128            // 2*FEA
#define MROWS    (NATOMS*MAXN)  // 49152
#define GCOEFF   (-0.5f / ((8.0f/63.0f)*(8.0f/63.0f)))

__device__ __forceinline__ float softplusf(float x) {
    return fmaxf(x, 0.0f) + log1pf(expf(-fabsf(x)));
}
__device__ __forceinline__ float sigmoidf_(float x) {
    return 1.0f / (1.0f + expf(-x));
}

// async global -> LDS 16B copy, tracked by ASYNCcnt (CDNA5 path)
__device__ __forceinline__ void async_copy_b128(void* lds_dst, const void* gsrc) {
    unsigned lds_addr = (unsigned)(uintptr_t)lds_dst;   // low 32 bits = LDS addr
    asm volatile("global_load_async_to_lds_b128 %0, %1, off"
                 :
                 : "v"(lds_addr), "v"(gsrc)
                 : "memory");
}
__device__ __forceinline__ void wait_asynccnt0() {
    asm volatile("s_wait_asynccnt 0x0" ::: "memory");
}

// -------------------------- fp32 [K,N] -> fp16 transposed [N,K] (for WMMA B)
__global__ void cvt_f16_transpose_kernel(const float* __restrict__ src,
                                         _Float16* __restrict__ dstT) {
    int t = blockIdx.x * 256 + threadIdx.x;
    if (t >= KDIM * NOUT) return;
    const int k = t / NOUT, n = t - k * NOUT;
    dstT[n * KDIM + k] = (_Float16)src[t];
}

// ------------------------------------------- softmax(species) @ emb_w + emb_b
// one wave per atom, 8 atoms per 256-thread block
__global__ void embed_kernel(const float* __restrict__ logits,
                             const float* __restrict__ emb_w,
                             const float* __restrict__ emb_b,
                             float* __restrict__ atom_fea) {
    __shared__ float sp[8][104];
    const int tid = threadIdx.x, wave = tid >> 5, lane = tid & 31;
    const int i = blockIdx.x * 8 + wave;
    const float* lg = logits + (size_t)i * SPECIES;

    float x[4]; float mx = -1e30f;
    #pragma unroll
    for (int t = 0; t < 4; ++t) {
        int s = lane + 32 * t;
        x[t] = (s < SPECIES) ? lg[s] : -1e30f;
        mx = fmaxf(mx, x[t]);
    }
    for (int off = 16; off >= 1; off >>= 1) mx = fmaxf(mx, __shfl_xor(mx, off));
    float e[4], sum = 0.f;
    #pragma unroll
    for (int t = 0; t < 4; ++t) {
        int s = lane + 32 * t;
        e[t] = (s < SPECIES) ? expf(x[t] - mx) : 0.f;
        sum += e[t];
    }
    for (int off = 16; off >= 1; off >>= 1) sum += __shfl_xor(sum, off);
    const float inv = 1.f / sum;
    #pragma unroll
    for (int t = 0; t < 4; ++t) {
        int s = lane + 32 * t;
        if (s < SPECIES) sp[wave][s] = e[t] * inv;
    }
    __syncthreads();
    #pragma unroll
    for (int cc = 0; cc < 2; ++cc) {
        int c = lane + 32 * cc;
        float acc = emb_b[c];
        for (int s = 0; s < SPECIES; ++s) acc += sp[wave][s] * emb_w[s * FEA + c];
        atom_fea[(size_t)i * FEA + c] = acc;
    }
}

// ---------------------------- min-image distances + top-12 (one wave per atom)
__global__ void topk_kernel(const float* __restrict__ lat,
                            const float* __restrict__ fracs,
                            int* __restrict__ nbr_idx,
                            float* __restrict__ nbr_dist) {
    __shared__ float sD[4][32 * MAXN];
    __shared__ int   sJ[4][32 * MAXN];
    const int tid = threadIdx.x, wave = tid >> 5, lane = tid & 31;
    const int i = blockIdx.x * 4 + wave;

    float L[9];
    #pragma unroll
    for (int t = 0; t < 9; ++t) L[t] = lat[t];
    const float fx = fracs[i * 3 + 0], fy = fracs[i * 3 + 1], fz = fracs[i * 3 + 2];

    float bd[MAXN]; int bj[MAXN];
    #pragma unroll
    for (int t = 0; t < MAXN; ++t) { bd[t] = 1e30f; bj[t] = -1; }

    for (int j0 = 0; j0 < NATOMS; j0 += 32) {
        const int j = j0 + lane;
        float dx = fx - fracs[j * 3 + 0];
        float dy = fy - fracs[j * 3 + 1];
        float dz = fz - fracs[j * 3 + 2];
        dx -= rintf(dx); dy -= rintf(dy); dz -= rintf(dz);   // round-half-even
        const float cx = dx * L[0] + dy * L[3] + dz * L[6];
        const float cy = dx * L[1] + dy * L[4] + dz * L[7];
        const float cz = dx * L[2] + dy * L[5] + dz * L[8];
        float d2 = cx * cx + cy * cy + cz * cz;
        if (j == i) d2 = 1e30f;                               // exclude self
        if (d2 < bd[MAXN - 1]) {                              // sorted insert
            bd[MAXN - 1] = d2; bj[MAXN - 1] = j;
            #pragma unroll
            for (int s = MAXN - 1; s > 0; --s)
                if (bd[s] < bd[s - 1]) {
                    float td = bd[s]; bd[s] = bd[s - 1]; bd[s - 1] = td;
                    int   tj = bj[s]; bj[s] = bj[s - 1]; bj[s - 1] = tj;
                }
        }
    }
    #pragma unroll
    for (int t = 0; t < MAXN; ++t) {
        sD[wave][lane * MAXN + t] = bd[t];
        sJ[wave][lane * MAXN + t] = bj[t];
    }
    __syncthreads();

    int p = 0;  // merge 32 sorted lists -> global 12 smallest
    for (int r = 0; r < MAXN; ++r) {
        const float cand = (p < MAXN) ? sD[wave][lane * MAXN + p] : 1e30f;
        const int   cj   = (p < MAXN) ? sJ[wave][lane * MAXN + p] : -1;
        float m = cand;
        for (int off = 16; off >= 1; off >>= 1) m = fminf(m, __shfl_xor(m, off));
        const unsigned long long bal = __ballot(cand == m) & 0xffffffffull;
        const int win = __ffsll(bal) - 1;
        const int jw = __shfl(cj, win);
        if (lane == 0) {
            nbr_idx [i * MAXN + r] = jw;
            nbr_dist[i * MAXN + r] = sqrtf(m);
        }
        if (lane == win) ++p;
    }
}

// --------- build concat matrix [MROWS,192] f16: [atom | nbr_atom | gaussian]
__global__ void build_total_kernel(const float* __restrict__ atom_fea,
                                   const int* __restrict__ nbr_idx,
                                   const float* __restrict__ nbr_dist,
                                   _Float16* __restrict__ total) {
    const int idx = blockIdx.x * 256 + threadIdx.x;
    if (idx >= MROWS * KDIM) return;
    const int r = idx / KDIM, c = idx - r * KDIM;
    const int i = r / MAXN;
    float v;
    if (c < FEA) {
        v = atom_fea[(size_t)i * FEA + c];
    } else if (c < 2 * FEA) {
        const int j = nbr_idx[r];
        v = atom_fea[(size_t)j * FEA + (c - FEA)];
    } else {
        const float d = nbr_dist[r];
        const float off = (8.0f / 63.0f) * (float)(c - 2 * FEA);
        const float t = d - off;
        v = expf(GCOEFF * t * t);
    }
    total[idx] = (_Float16)v;
}

// ------------------- [MROWS,192] @ [192,128] with v_wmma_f32_16x16x32_f16
// block: 256 thr (8 waves) -> 32 rows x 128 cols; wave w owns col-tile w,
// computes two 16x16 row tiles; K = 192 = 6 chunks of 32.
// Operands staged global->LDS via ASYNCcnt-tracked async b128 copies; weight
// pre-transposed so each lane's A/B fragments are contiguous ds_load_b128.
__global__ void gemm_wmma_kernel(const _Float16* __restrict__ A,
                                 const _Float16* __restrict__ BwT,
                                 float* __restrict__ Z) {
    __shared__ _Float16 sA [32 * KDIM];    // 12 KB
    __shared__ _Float16 sBt[NOUT * KDIM];  // 48 KB, [n][k]
    const int tid = threadIdx.x;
    const int wave = tid >> 5, lane = tid & 31;
    const int rowBase = blockIdx.x * 32;

    {   // async-stage A tile (32x192 halves = 768 x 16B)
        const char* src = (const char*)(A + (size_t)rowBase * KDIM);
        char* dst = (char*)sA;
        for (int t = tid; t < 768; t += 256)
            async_copy_b128(dst + t * 16, src + t * 16);
    }
    {   // async-stage transposed weight (128x192 halves = 3072 x 16B)
        const char* src = (const char*)BwT;
        char* dst = (char*)sBt;
        for (int t = tid; t < 3072; t += 256)
            async_copy_b128(dst + t * 16, src + t * 16);
    }
    // L2 prefetch of the next block's A tile while we wait
    if (rowBase + 32 < MROWS)
        __builtin_prefetch((const char*)(A + (size_t)(rowBase + 32) * KDIM) + tid * 48, 0, 1);
    wait_asynccnt0();
    __syncthreads();

    const int n0 = wave * 16;
    const int l16 = lane & 15;
    const int h = lane >> 4;
    const _Float16* bRow = sBt + (n0 + l16) * KDIM;   // column n of B

    #pragma unroll
    for (int rt = 0; rt < 2; ++rt) {
        v8f acc = {};
        const _Float16* aRow = sA + (rt * 16 + l16) * KDIM;
        #pragma unroll
        for (int kk = 0; kk < 6; ++kk) {
            // A 16x32 f16, contiguous k order shared with B: lane group h holds K = kk*32+16h .. +15 of row m
            v16h a = *(const v16h*)(aRow + kk * 32 + 16 * h);
            // B 32x16 f16, contiguous k order shared with A: lane group h holds K = kk*32+16h .. +15 of col n
            v16h b = *(const v16h*)(bRow + kk * 32 + 16 * h);
            acc = __builtin_amdgcn_wmma_f32_16x16x32_f16(
                      false, a, false, b, (short)0, acc, false, false);
        }
        // C/D layout: VGPR j -> row j + 8*h, col = lane%16
        float* zBase = Z + (size_t)(rowBase + rt * 16) * NOUT;
        #pragma unroll
        for (int j = 0; j < 8; ++j)
            zBase[(j + 8 * h) * NOUT + n0 + l16] = acc[j];
    }
}

// ---- +bias, layernorm, split, sigmoid*softplus, sum over nbrs, softplus(res)
// one 128-thread block (4 waves) per atom; each wave handles 3 neighbor rows
__global__ void ln_act_agg_kernel(const float* __restrict__ Z,
                                  const float* __restrict__ bias,
                                  const float* __restrict__ gamma,
                                  const float* __restrict__ beta,
                                  const float* __restrict__ atom_in,
                                  float* __restrict__ atom_out) {
    __shared__ float sacc[FEA];
    const int i = blockIdx.x;
    const int tid = threadIdx.x, wave = tid >> 5, lane = tid & 31;
    if (tid < FEA) sacc[tid] = 0.f;
    __syncthreads();

    for (int mm = wave; mm < MAXN; mm += 4) {
        const float* zr = Z + (size_t)(i * MAXN + mm) * NOUT;
        float v[4];
        #pragma unroll
        for (int t = 0; t < 4; ++t) { int c = lane + 32 * t; v[t] = zr[c] + bias[c]; }
        float s = v[0] + v[1] + v[2] + v[3];
        for (int off = 16; off >= 1; off >>= 1) s += __shfl_xor(s, off);
        const float mean = s * (1.0f / NOUT);
        float q = 0.f;
        #pragma unroll
        for (int t = 0; t < 4; ++t) { float d = v[t] - mean; q += d * d; }
        for (int off = 16; off >= 1; off >>= 1) q += __shfl_xor(q, off);
        const float inv = rsqrtf(q * (1.0f / NOUT) + 1e-5f);
        float y[4];
        #pragma unroll
        for (int t = 0; t < 4; ++t) {
            int c = lane + 32 * t;
            y[t] = (v[t] - mean) * inv * gamma[c] + beta[c];
        }
        #pragma unroll
        for (int t = 0; t < 2; ++t) {        // filter col c pairs core col c+64
            int c = lane + 32 * t;
            atomicAdd(&sacc[c], sigmoidf_(y[t]) * softplusf(y[t + 2]));
        }
    }
    __syncthreads();
    if (tid < FEA)
        atom_out[(size_t)i * FEA + tid] =
            softplusf(atom_in[(size_t)i * FEA + tid] + sacc[tid]);
}

// ------------------------------- occupancy-weighted pooling + final fc -> [1]
__global__ void pool_kernel(const float* __restrict__ atom_fea,
                            const float* __restrict__ occ,
                            const float* __restrict__ fc_w,
                            const float* __restrict__ fc_b,
                            float* __restrict__ out) {
    __shared__ float sfeat[FEA];
    __shared__ float socc, sdot;
    const int tid = threadIdx.x;
    if (tid < FEA) sfeat[tid] = 0.f;
    if (tid == 0) { socc = 0.f; sdot = 0.f; }
    __syncthreads();
    const int c = tid & 63, g = tid >> 6;   // 4 groups of 64 cols
    float acc = 0.f, osum = 0.f;
    for (int i = g; i < NATOMS; i += 4) {
        const float w = sigmoidf_(occ[i]);
        acc += atom_fea[(size_t)i * FEA + c] * w;
        if (c == 0) osum += w;
    }
    atomicAdd(&sfeat[c], acc);
    if (c == 0) atomicAdd(&socc, osum);
    __syncthreads();
    if (tid < FEA) atomicAdd(&sdot, sfeat[tid] / (socc + 1e-6f) * fc_w[tid]);
    __syncthreads();
    if (tid == 0) out[0] = sdot + fc_b[0];
}

extern "C" void kernel_launch(void* const* d_in, const int* in_sizes, int n_in,
                              void* d_out, int out_size, void* d_ws, size_t ws_size,
                              hipStream_t stream) {
    const float* lat   = (const float*)d_in[0];
    const float* fracs = (const float*)d_in[1];
    const float* slog  = (const float*)d_in[2];
    const float* occ   = (const float*)d_in[3];
    const float* emb_w = (const float*)d_in[4];
    const float* emb_b = (const float*)d_in[5];
    const float* w1    = (const float*)d_in[6];
    const float* b1    = (const float*)d_in[7];
    const float* g1    = (const float*)d_in[8];
    const float* be1   = (const float*)d_in[9];
    const float* w2    = (const float*)d_in[10];
    const float* b2    = (const float*)d_in[11];
    const float* g2    = (const float*)d_in[12];
    const float* be2   = (const float*)d_in[13];
    const float* fc_w  = (const float*)d_in[14];
    const float* fc_b  = (const float*)d_in[15];
    float* out = (float*)d_out;

    char* ws = (char*)d_ws;
    float*    fea0  = (float*)   (ws + 0x0000000);               // 4096*64 f32
    float*    fea1  = (float*)   (ws + 0x0100000);               // 4096*64 f32
    int*      nidx  = (int*)     (ws + 0x0200000);               // 4096*12 i32
    float*    ndist = (float*)   (ws + 0x0230000);               // 4096*12 f32
    _Float16* w1hT  = (_Float16*)(ws + 0x0260000);               // 128*192 f16
    _Float16* w2hT  = (_Float16*)(ws + 0x026C000);               // 128*192 f16
    _Float16* total = (_Float16*)(ws + 0x0278000);               // 49152*192 f16
    float*    z     = (float*)   (ws + 0x1478000);               // 49152*128 f32

    const int nw = KDIM * NOUT;                                  // 24576
    cvt_f16_transpose_kernel<<<(nw + 255) / 256, 256, 0, stream>>>(w1, w1hT);
    cvt_f16_transpose_kernel<<<(nw + 255) / 256, 256, 0, stream>>>(w2, w2hT);

    embed_kernel<<<NATOMS / 8, 256, 0, stream>>>(slog, emb_w, emb_b, fea0);
    topk_kernel<<<NATOMS / 4, 128, 0, stream>>>(lat, fracs, nidx, ndist);

    const int totElems = MROWS * KDIM;
    // ---- layer 1
    build_total_kernel<<<(totElems + 255) / 256, 256, 0, stream>>>(fea0, nidx, ndist, total);
    gemm_wmma_kernel<<<MROWS / 32, 256, 0, stream>>>(total, w1hT, z);
    ln_act_agg_kernel<<<NATOMS, 128, 0, stream>>>(z, b1, g1, be1, fea0, fea1);
    // ---- layer 2
    build_total_kernel<<<(totElems + 255) / 256, 256, 0, stream>>>(fea1, nidx, ndist, total);
    gemm_wmma_kernel<<<MROWS / 32, 256, 0, stream>>>(total, w2hT, z);
    ln_act_agg_kernel<<<NATOMS, 128, 0, stream>>>(z, b2, g2, be2, fea1, fea0);

    pool_kernel<<<1, 256, 0, stream>>>(fea0, occ, fc_w, fc_b, out);
}
